// GestureCNNLSTM_ManualUnroll_23158463660372
// MI455X (gfx1250) — compile-verified, matrix operand order checked by reading, not run on hardware
//
#include <hip/hip_runtime.h>
#include <hip/hip_bf16.h>

// ---------------------------------------------------------------------------
// GestureCNNLSTM on MI455X (gfx1250, wave32, WMMA).
//  - conv stack: VALU kernel, one 64-thread block per image, LDS staging
//  - LSTM: fused [feat_t | h] x [W_ih;W_hh]^T per step with v_wmma_f32_16x16x32_bf16
//    Weight tiles staged per-block into LDS with GLOBAL_LOAD_ASYNC_TO_LDS_B128
//    (ASYNCcnt + s_wait_asynccnt), inner loop = ds_load_b128 + WMMA.
//    512 stream-ordered step launches (double-buffered h), gate math fused.
//  - classifier: single-block WMMA kernel
// ---------------------------------------------------------------------------

typedef __bf16 bf16;
typedef bf16  v16bf __attribute__((ext_vector_type(16)));
typedef float v8f   __attribute__((ext_vector_type(8)));

#define T_STEPS 512
#define BATCH   128
#define HID     256
#define FEAT    192
#define KTOT    448      // FEAT + HID, = 14 * 32
#define GATES   1024     // 4 * HID
#define TILE_BYTES 14336 // one ntile: 14 kt * 32 lanes * 16 bf16 * 2B

union Frag { uint4 u[2]; v16bf v; };

__device__ __forceinline__ v8f zero8() {
    v8f z;
    #pragma unroll
    for (int i = 0; i < 8; ++i) z[i] = 0.0f;
    return z;
}

__device__ __forceinline__ float sigf(float x) { return 1.0f / (1.0f + __expf(-x)); }

// ---------------------------------------------------------------------------
// Init: bias = b_ih + b_hh; zero h double-buffers (bf16) and c state (f32)
// ---------------------------------------------------------------------------
__global__ void init_kernel(const float* __restrict__ b_ih, const float* __restrict__ b_hh,
                            float* __restrict__ bias, bf16* __restrict__ hA,
                            bf16* __restrict__ hB, float* __restrict__ cst) {
    int idx = blockIdx.x * blockDim.x + threadIdx.x;
    if (idx < GATES) bias[idx] = b_ih[idx] + b_hh[idx];
    if (idx < BATCH * HID) {
        hA[idx]  = (bf16)0.0f;
        hB[idx]  = (bf16)0.0f;
        cst[idx] = 0.0f;
    }
}

// ---------------------------------------------------------------------------
// Pack W = [W_ih | W_hh] (1024 x 448, fp32) into bf16 B-fragment layout:
//   wpack[((ntile*14 + kt)*32 + lane)*16 + e]
//   lane<16:  N = ntile*16+lane,    e -> K = kt*32 + {0..7,16..23}
//   lane>=16: N = ntile*16+lane-16, e -> K = kt*32 + {8..15,24..31}
// so each lane loads its 16 bf16 fragment elements as two b128s.
// ---------------------------------------------------------------------------
__global__ void pack_w_kernel(const float* __restrict__ W_ih, const float* __restrict__ W_hh,
                              bf16* __restrict__ wpack) {
    int idx = blockIdx.x * blockDim.x + threadIdx.x;
    const int total = 64 * 14 * 32 * 16;   // 458752 = 1024*448
    if (idx >= total) return;
    int e     = idx & 15;
    int lane  = (idx >> 4) & 31;
    int kt    = (idx >> 9) % 14;
    int ntile = idx / (14 * 32 * 16);
    int n    = ntile * 16 + (lane & 15);
    int kidx = (e < 8 ? e : e + 8) + ((lane >> 4) << 3);
    int k    = kt * 32 + kidx;
    float v  = (k < FEAT) ? W_ih[(size_t)n * FEAT + k]
                          : W_hh[(size_t)n * HID + (k - FEAT)];
    wpack[idx] = (bf16)v;
}

// ---------------------------------------------------------------------------
// Conv stack: one image per 64-thread block.
// conv1(1->16,3x3,pad1)+relu+pool2 : 13x10 -> 16x6x5
// conv2(16->32,3x3,pad1)+relu+pool2:  6x5  -> 32x3x2 = 192 feats (bf16 out)
// ---------------------------------------------------------------------------
__global__ __launch_bounds__(64) void conv_feat_kernel(
    const float* __restrict__ x,
    const float* __restrict__ c1w, const float* __restrict__ c1b,
    const float* __restrict__ c2w, const float* __restrict__ c2b,
    bf16* __restrict__ featbf) {
    __shared__ float s_in[130];
    __shared__ float s_c1w[144];
    __shared__ float s_c1b[16];
    __shared__ float s_c2w[4608];
    __shared__ float s_c2b[32];
    __shared__ float s_p1[480];   // 16 x 6 x 5 pooled conv1

    const int tid = threadIdx.x;
    const size_t img = blockIdx.x;

    for (int i = tid; i < 130;  i += 64) s_in[i]  = x[img * 130 + i];
    for (int i = tid; i < 144;  i += 64) s_c1w[i] = c1w[i];
    for (int i = tid; i < 16;   i += 64) s_c1b[i] = c1b[i];
    for (int i = tid; i < 4608; i += 64) s_c2w[i] = c2w[i];
    for (int i = tid; i < 32;   i += 64) s_c2b[i] = c2b[i];
    __syncthreads();

    // conv1 + relu + maxpool2 (compute pooled outputs directly)
    for (int p = tid; p < 480; p += 64) {
        int c = p / 30, rem = p % 30, ph = rem / 5, pw = rem % 5;
        float m = 0.0f;   // relu >= 0, so 0 is a valid identity for the max
        #pragma unroll
        for (int dy = 0; dy < 2; ++dy)
        #pragma unroll
        for (int dx = 0; dx < 2; ++dx) {
            int y = 2 * ph + dy, xx = 2 * pw + dx;
            float acc = s_c1b[c];
            #pragma unroll
            for (int ky = 0; ky < 3; ++ky)
            #pragma unroll
            for (int kx = 0; kx < 3; ++kx) {
                int iy = y + ky - 1, ix = xx + kx - 1;
                if (iy >= 0 && iy < 13 && ix >= 0 && ix < 10)
                    acc += s_c1w[c * 9 + ky * 3 + kx] * s_in[iy * 10 + ix];
            }
            float r = acc > 0.0f ? acc : 0.0f;
            m = r > m ? r : m;
        }
        s_p1[p] = m;
    }
    __syncthreads();

    // conv2 + relu + maxpool2 -> 192 features, channel-major flatten
    for (int p = tid; p < 192; p += 64) {
        int c = p / 6, rem = p % 6, ph = rem / 2, pw = rem % 2;
        float m = 0.0f;
        #pragma unroll
        for (int dy = 0; dy < 2; ++dy)
        #pragma unroll
        for (int dx = 0; dx < 2; ++dx) {
            int y = 2 * ph + dy, xx = 2 * pw + dx;   // within 6x5
            float acc = s_c2b[c];
            for (int ic = 0; ic < 16; ++ic) {
                #pragma unroll
                for (int ky = 0; ky < 3; ++ky)
                #pragma unroll
                for (int kx = 0; kx < 3; ++kx) {
                    int iy = y + ky - 1, ix = xx + kx - 1;
                    if (iy >= 0 && iy < 6 && ix >= 0 && ix < 5)
                        acc += s_c2w[((c * 16 + ic) * 3 + ky) * 3 + kx] *
                               s_p1[ic * 30 + iy * 5 + ix];
                }
            }
            float r = acc > 0.0f ? acc : 0.0f;
            m = r > m ? r : m;
        }
        featbf[img * 192 + p] = (bf16)m;
    }
}

// ---------------------------------------------------------------------------
// One LSTM step: gates = [feat_t | h_in] (128x448) x Wpack^T (448x1024) + bias
// 16 blocks (one per 16-col slice of H), 8 waves/block. Wave w = M-tile w,
// loops over the 4 gate N-tiles; K = 14 steps of 32 (bf16 WMMA).
//
// The block's 4 weight tiles (56 KB) are staged into LDS once via
// GLOBAL_LOAD_ASYNC_TO_LDS_B128 (all 8 waves would otherwise fetch identical
// B fragments from L2 -> 8x redundant VMEM). Inner loop: ds_load_b128 (LDS
// broadcast, conflict-free since all waves read the same address) + WMMA.
// Epilogue fuses sigmoid/tanh gate math, c update, and bf16 h_out write.
// ---------------------------------------------------------------------------
__global__ __launch_bounds__(256) void lstm_step_kernel(
    const bf16* __restrict__ featbf, const bf16* __restrict__ wpack,
    const float* __restrict__ bias, const bf16* __restrict__ hin,
    bf16* __restrict__ hout, float* __restrict__ cst, int t) {
    __shared__ __align__(16) unsigned char s_w[4 * TILE_BYTES];   // 56 KB

    const int tid     = threadIdx.x;
    const int lane    = threadIdx.x & 31;
    const int wave    = threadIdx.x >> 5;   // 0..7 -> M tile
    const int j       = blockIdx.x;         // 0..15 -> 16-col slice of H
    const int halfsel = lane >> 4;          // which K-half this lane holds
    const int nrow    = lane & 15;

    // ---- async-stage the 4 gate weight tiles for this column slice ----
    {
        const unsigned lds_base = (unsigned)(unsigned long long)(&s_w[0]);
        #pragma unroll
        for (int q = 0; q < 4; ++q) {
            const char* src = (const char*)wpack + (size_t)(q * 16 + j) * TILE_BYTES;
            for (int i = tid; i < TILE_BYTES / 16; i += 256) {
                unsigned           ldsoff = lds_base + q * TILE_BYTES + i * 16;
                unsigned long long gaddr  = (unsigned long long)src + (unsigned long long)i * 16;
                asm volatile("global_load_async_to_lds_b128 %0, %1, off"
                             :: "v"(ldsoff), "v"(gaddr) : "memory");
            }
        }
        asm volatile("s_wait_asynccnt 0" ::: "memory");
    }
    __syncthreads();

    v8f acc[4];
    #pragma unroll
    for (int q = 0; q < 4; ++q) acc[q] = zero8();

    const int arow = wave * 16 + nrow;      // batch row for A fragment

    #pragma unroll 2
    for (int kt = 0; kt < 14; ++kt) {
        // --- A fragment: two contiguous 16B groups from feat row or h row ---
        Frag af;
        int g0 = kt * 32 + (halfsel << 3);
        const bf16* asrc;
        int off;
        if (kt < 6) { asrc = featbf + ((size_t)arow * T_STEPS + t) * FEAT; off = g0; }
        else        { asrc = hin + (size_t)arow * HID;                     off = g0 - FEAT; }
        af.u[0] = *(const uint4*)(asrc + off);
        af.u[1] = *(const uint4*)(asrc + off + 16);

        // --- 4 gate B fragments from LDS + WMMAs ---
        #pragma unroll
        for (int q = 0; q < 4; ++q) {
            const uint4* wlds =
                (const uint4*)(s_w + q * TILE_BYTES + (kt * 32 + lane) * 32);
            Frag bfr;
            bfr.u[0] = wlds[0];
            bfr.u[1] = wlds[1];
            acc[q] = __builtin_amdgcn_wmma_f32_16x16x32_bf16(
                false, af.v, false, bfr.v, (short)0, acc[q], false, false);
        }
    }

    // --- fused gate nonlinearity + state update ---
    const int hcol = j * 16 + nrow;
    #pragma unroll
    for (int v = 0; v < 8; ++v) {
        int row = wave * 16 + v + (halfsel << 3);
        float iv = sigf(acc[0][v] + bias[hcol]);
        float fv = sigf(acc[1][v] + bias[256 + hcol]);
        float gv = tanhf(acc[2][v] + bias[512 + hcol]);
        float ov = sigf(acc[3][v] + bias[768 + hcol]);
        size_t idx = (size_t)row * HID + hcol;
        float cc = fv * cst[idx] + iv * gv;
        cst[idx]  = cc;
        hout[idx] = (bf16)(ov * tanhf(cc));
    }
}

// ---------------------------------------------------------------------------
// Classifier: logits(128x64) = hT(128x256) x cls_w^T + cls_b, one block.
// ---------------------------------------------------------------------------
__global__ __launch_bounds__(256) void cls_kernel(
    const bf16* __restrict__ hfin, const float* __restrict__ cls_w,
    const float* __restrict__ cls_b, float* __restrict__ out) {
    const int lane = threadIdx.x & 31;
    const int wave = threadIdx.x >> 5;
    const int half = lane >> 4;
    const int nrow = lane & 15;
    const int arow = wave * 16 + nrow;

    for (int ntile = 0; ntile < 4; ++ntile) {
        v8f acc = zero8();
        const int ncol = ntile * 16 + nrow;
        const float* wrow = cls_w + (size_t)ncol * HID;
        #pragma unroll 2
        for (int kt = 0; kt < 8; ++kt) {
            Frag af;
            int g0 = kt * 32 + (half << 3);
            const bf16* asrc = hfin + (size_t)arow * HID;
            af.u[0] = *(const uint4*)(asrc + g0);
            af.u[1] = *(const uint4*)(asrc + g0 + 16);
            Frag bfr;
            #pragma unroll
            for (int e = 0; e < 16; ++e) {
                int kidx = (e < 8 ? e : e + 8) + (half << 3);
                bfr.v[e] = (bf16)wrow[kt * 32 + kidx];
            }
            acc = __builtin_amdgcn_wmma_f32_16x16x32_bf16(
                false, af.v, false, bfr.v, (short)0, acc, false, false);
        }
        #pragma unroll
        for (int v = 0; v < 8; ++v) {
            int row = wave * 16 + v + (half << 3);
            out[(size_t)row * 64 + ncol] = acc[v] + cls_b[ncol];
        }
    }
}

// ---------------------------------------------------------------------------
// Launcher. Workspace layout (bytes, all offsets 256-aligned):
//   featbf : 0          .. 25,165,824   (65536*192 bf16)
//   wpack  : 25,165,824 .. 26,083,328   (1024*448 bf16)
//   bias   : 26,083,328 .. 26,087,424   (1024 f32)
//   hA     : 26,087,424 .. 26,152,960   (128*256 bf16)
//   hB     : 26,152,960 .. 26,218,496   (128*256 bf16)
//   cstate : 26,218,496 .. 26,349,568   (128*256 f32)
// ---------------------------------------------------------------------------
extern "C" void kernel_launch(void* const* d_in, const int* in_sizes, int n_in,
                              void* d_out, int out_size, void* d_ws, size_t ws_size,
                              hipStream_t stream) {
    const float* x     = (const float*)d_in[0];
    const float* c1w   = (const float*)d_in[1];
    const float* c1b   = (const float*)d_in[2];
    const float* c2w   = (const float*)d_in[3];
    const float* c2b   = (const float*)d_in[4];
    const float* W_ih  = (const float*)d_in[5];
    const float* b_ih  = (const float*)d_in[6];
    const float* W_hh  = (const float*)d_in[7];
    const float* b_hh  = (const float*)d_in[8];
    const float* cls_w = (const float*)d_in[9];
    const float* cls_b = (const float*)d_in[10];
    float* out = (float*)d_out;

    char* ws = (char*)d_ws;
    bf16*  featbf = (bf16*)(ws);
    bf16*  wpack  = (bf16*)(ws + 25165824);
    float* bias   = (float*)(ws + 26083328);
    bf16*  hA     = (bf16*)(ws + 26087424);
    bf16*  hB     = (bf16*)(ws + 26152960);
    float* cst    = (float*)(ws + 26218496);

    init_kernel<<<(BATCH * HID + 255) / 256, 256, 0, stream>>>(b_ih, b_hh, bias, hA, hB, cst);
    pack_w_kernel<<<(64 * 14 * 32 * 16 + 255) / 256, 256, 0, stream>>>(W_ih, W_hh, wpack);
    conv_feat_kernel<<<BATCH * T_STEPS, 64, 0, stream>>>(x, c1w, c1b, c2w, c2b, featbf);

    bf16* hin = hA;
    bf16* hout = hB;
    for (int t = 0; t < T_STEPS; ++t) {
        lstm_step_kernel<<<16, 256, 0, stream>>>(featbf, wpack, bias, hin, hout, cst, t);
        bf16* tmp = hin; hin = hout; hout = tmp;
    }

    cls_kernel<<<1, 256, 0, stream>>>(hin, cls_w, cls_b, out);
}
